// MambaBlock_10187662426898
// MI455X (gfx1250) — compile-verified, matrix operand order checked by reading, not run on hardware
//
#include <hip/hip_runtime.h>
#include <hip/hip_bf16.h>

// ---------------------------------------------------------------------------
// Types for CDNA5 WMMA
// ---------------------------------------------------------------------------
typedef __bf16 bf16;
typedef bf16  v16bf __attribute__((ext_vector_type(16)));
typedef bf16  v8bf  __attribute__((ext_vector_type(8)));
typedef float v8f   __attribute__((ext_vector_type(8)));
typedef int   v4i   __attribute__((ext_vector_type(4)));
typedef __attribute__((address_space(1))) v4i* as1_v4i;
typedef __attribute__((address_space(3))) v4i* as3_v4i;

#define D_MODEL   1024
#define D_INNER   2048
#define D_STATE   128
#define NHEADS    32
#define HEADDIM   64
#define CONV_DIM  2304           // D_INNER + 2*D_STATE
#define N_AY      3072           // 3*D_MODEL
#define N_ZX      4384           // D_IN_PROJ
#define N_ZX_PAD  4480           // padded to multiple of 128
#define BATCH     8
#define SEQ       1024
#define ROWS      (BATCH*SEQ)    // 8192

// Async global->LDS staging (gfx1250): guard on builtin availability.
#if defined(__has_builtin)
#if __has_builtin(__builtin_amdgcn_global_load_async_to_lds_b128)
#define ASYNC_COPY 1
#endif
#endif
#ifndef ASYNC_COPY
#define ASYNC_COPY 0
#endif

__device__ __forceinline__ float silu_f(float x) { return x / (1.0f + __expf(-x)); }
__device__ __forceinline__ bf16  to_bf16(float f) { return (bf16)f; }

#if ASYNC_COPY
// Issue one 16-byte async global->LDS copy (ASYNCcnt-tracked).
__device__ __forceinline__ void async_copy_b128(const void* g, void* l) {
    __builtin_amdgcn_global_load_async_to_lds_b128(
        (as1_v4i)const_cast<void*>(g), (as3_v4i)l, 0, 0);
}
#endif

// ---------------------------------------------------------------------------
// WMMA bf16 GEMM: C[M,N](f32) = A[M,K](bf16 row-major) * Bt[N,K](bf16 row-major)^T
//                              (+ bias[n] if bias != nullptr)
// Block: 256 threads = 8 wave32s, block tile 128x128, wave tile 64x32.
// K-step 64, double-buffered LDS staging via GLOBAL_LOAD_ASYNC_TO_LDS_B128
// (ASYNCcnt-gated), rows padded to 144B to avoid LDS bank conflicts.
// A-frag per lane: row m = lane&15, 16B at K-offsets half*8 and 16+half*8
// (CDNA5 16-bit A layout).  B-frag identical on the N-major transposed weights.
// C/D layout: VGPR r holds row (r + half*8), col = lane&15.
// M, N multiples of 128; K multiple of 64 (guaranteed by padding).
// ---------------------------------------------------------------------------
#define KSTEP 64
#define SROW  72   // LDS row stride in bf16 elements (144 bytes, 16B-aligned)

__device__ __forceinline__ v16bf load_frag16(const bf16* __restrict__ rowk, int half) {
    const v8bf lo = *reinterpret_cast<const v8bf*>(rowk + half * 8);
    const v8bf hi = *reinterpret_cast<const v8bf*>(rowk + 16 + half * 8);
    v16bf r;
#pragma unroll
    for (int i = 0; i < 8; ++i) { r[i] = lo[i]; r[i + 8] = hi[i]; }
    return r;
}

__global__ __launch_bounds__(256) void wmma_gemm_bf16(
    const bf16* __restrict__ A, const bf16* __restrict__ Bt,
    float* __restrict__ C, const float* __restrict__ bias,
    int K, int lda, int ldb, int ldc)
{
    const int tid    = threadIdx.x;
    const int lid    = tid & 31;
    const int wid    = tid >> 5;     // 0..7
    const int waveM  = wid >> 2;     // 0..1
    const int waveN  = wid & 3;      // 0..3
    const int half   = lid >> 4;     // 0..1
    const int lane16 = lid & 15;

    const int bm0 = blockIdx.y * 128;
    const int bn0 = blockIdx.x * 128;

    v8f acc[4][2] = {};

#if ASYNC_COPY
    __shared__ __align__(16) bf16 As[2][128 * SROW];
    __shared__ __align__(16) bf16 Bs[2][128 * SROW];

    // Stage one 128x64 A tile + 128x64 B tile: 1024 b128 transfers each,
    // 256 threads -> 4 (A) + 4 (B) async instructions per thread per stage.
    auto issue_stage = [&](int kk, int sel) {
#pragma unroll
        for (int rep = 0; rep < 4; ++rep) {
            const int i   = tid + rep * 256;   // 0..1023
            const int row = i >> 3;
            const int ch  = i & 7;             // 16B chunk within 128B row
            const bf16* gA = A  + (size_t)(bm0 + row) * lda + kk + ch * 8;
            const bf16* gB = Bt + (size_t)(bn0 + row) * ldb + kk + ch * 8;
            async_copy_b128(gA, &As[sel][row * SROW + ch * 8]);
            async_copy_b128(gB, &Bs[sel][row * SROW + ch * 8]);
        }
    };

    const int nk = K / KSTEP;
    issue_stage(0, 0);
    for (int ks = 0; ks < nk; ++ks) {
        if (ks + 1 < nk) {
            issue_stage((ks + 1) * KSTEP, (ks + 1) & 1);
            // 8 async instrs outstanding for stage ks+1; <=8 means stage ks landed.
            asm volatile("s_wait_asynccnt 0x8" ::: "memory");
        } else {
            asm volatile("s_wait_asynccnt 0x0" ::: "memory");
        }
        __syncthreads();

        const bf16* Abuf = &As[ks & 1][0];
        const bf16* Bbuf = &Bs[ks & 1][0];
#pragma unroll
        for (int k2 = 0; k2 < KSTEP; k2 += 32) {
            v16bf afrag[4], bfrag[2];
#pragma unroll
            for (int mt = 0; mt < 4; ++mt)
                afrag[mt] = load_frag16(Abuf + (waveM * 64 + mt * 16 + lane16) * SROW + k2, half);
#pragma unroll
            for (int nt = 0; nt < 2; ++nt)
                bfrag[nt] = load_frag16(Bbuf + (waveN * 32 + nt * 16 + lane16) * SROW + k2, half);
#pragma unroll
            for (int mt = 0; mt < 4; ++mt)
#pragma unroll
                for (int nt = 0; nt < 2; ++nt)
                    acc[mt][nt] = __builtin_amdgcn_wmma_f32_16x16x32_bf16(
                        false, afrag[mt], false, bfrag[nt],
                        (short)0, acc[mt][nt], false, false);
        }
        __syncthreads();
    }
#else
    // Fallback: direct global fragment loads (L2-resident operands).
    const int bm = bm0 + waveM * 64;
    const int bn = bn0 + waveN * 32;
    for (int kk = 0; kk < K; kk += 32) {
        v16bf afrag[4], bfrag[2];
#pragma unroll
        for (int mt = 0; mt < 4; ++mt)
            afrag[mt] = load_frag16(A + (size_t)(bm + mt * 16 + lane16) * lda + kk, half);
#pragma unroll
        for (int nt = 0; nt < 2; ++nt)
            bfrag[nt] = load_frag16(Bt + (size_t)(bn + nt * 16 + lane16) * ldb + kk, half);
#pragma unroll
        for (int mt = 0; mt < 4; ++mt)
#pragma unroll
            for (int nt = 0; nt < 2; ++nt)
                acc[mt][nt] = __builtin_amdgcn_wmma_f32_16x16x32_bf16(
                    false, afrag[mt], false, bfrag[nt],
                    (short)0, acc[mt][nt], false, false);
    }
#endif

    const int bmw = bm0 + waveM * 64;
    const int bnw = bn0 + waveN * 32;
#pragma unroll
    for (int mt = 0; mt < 4; ++mt)
#pragma unroll
        for (int nt = 0; nt < 2; ++nt) {
            const int col = bnw + nt * 16 + lane16;
            const float bv = bias ? bias[col] : 0.0f;
#pragma unroll
            for (int r = 0; r < 8; ++r) {
                const int row = bmw + mt * 16 + half * 8 + r;
                C[(size_t)row * ldc + col] = acc[mt][nt][r] + bv;
            }
        }
}

// ---------------------------------------------------------------------------
// Weight prep: Wt[n][k] = (bf16) W[k][n], rows padded with zeros up to Npad.
// ---------------------------------------------------------------------------
__global__ void transpose_convert(const float* __restrict__ W, bf16* __restrict__ out,
                                  int K, int N, int Npad, int total)
{
    int i = blockIdx.x * blockDim.x + threadIdx.x;
    if (i >= total) return;
    int n = i / K;
    int k = i - n * K;
    out[i] = (n < N) ? to_bf16(W[(size_t)k * N + n]) : to_bf16(0.0f);
}

// a_y = (bf16) silu(y)
__global__ void silu_cvt(const float* __restrict__ y, bf16* __restrict__ out, int total)
{
    int i = blockIdx.x * blockDim.x + threadIdx.x;
    if (i >= total) return;
    out[i] = to_bf16(silu_f(y[i]));
}

// at = silu(t) @ w_ada + b_ada   (8 x 3072, K=1024) — tiny, VALU dot products.
__global__ __launch_bounds__(256) void at_kernel(const float* __restrict__ t,
                                                 const float* __restrict__ w_ada,
                                                 const float* __restrict__ b_ada,
                                                 float* __restrict__ at)
{
    __shared__ float sT[BATCH * D_MODEL];
    for (int i = threadIdx.x; i < BATCH * D_MODEL; i += 256) sT[i] = silu_f(t[i]);
    __syncthreads();
    const int j = blockIdx.x * 256 + threadIdx.x;
    if (j >= N_AY) return;
    float acc[BATCH];
#pragma unroll
    for (int b = 0; b < BATCH; ++b) acc[b] = b_ada[j];
    for (int k = 0; k < D_MODEL; ++k) {
        const float wv = w_ada[(size_t)k * N_AY + j];
#pragma unroll
        for (int b = 0; b < BATCH; ++b) acc[b] += sT[b * D_MODEL + k] * wv;
    }
#pragma unroll
    for (int b = 0; b < BATCH; ++b) at[(size_t)b * N_AY + j] = acc[b];
}

// mod = LN(x) * (1 + scale) + shift, output bf16 (GEMM operand).
__global__ __launch_bounds__(256) void mod_kernel(const float* __restrict__ x,
                                                  const float* __restrict__ Cay,
                                                  const float* __restrict__ at,
                                                  bf16* __restrict__ mod)
{
    const int row = blockIdx.x;          // 0..8191
    const int b   = row >> 10;
    const int tid = threadIdx.x;
    __shared__ float red[256];

    float loc[4];
    float s = 0.0f;
#pragma unroll
    for (int i = 0; i < 4; ++i) {
        loc[i] = x[(size_t)row * D_MODEL + tid + i * 256];
        s += loc[i];
    }
    red[tid] = s; __syncthreads();
    for (int st = 128; st > 0; st >>= 1) { if (tid < st) red[tid] += red[tid + st]; __syncthreads(); }
    const float mu = red[0] * (1.0f / D_MODEL);
    __syncthreads();

    float v = 0.0f;
#pragma unroll
    for (int i = 0; i < 4; ++i) { const float d = loc[i] - mu; v += d * d; }
    red[tid] = v; __syncthreads();
    for (int st = 128; st > 0; st >>= 1) { if (tid < st) red[tid] += red[tid + st]; __syncthreads(); }
    const float rsig = rsqrtf(red[0] * (1.0f / D_MODEL) + 1e-6f);

#pragma unroll
    for (int i = 0; i < 4; ++i) {
        const int c = tid + i * 256;
        const float shift = Cay[(size_t)row * N_AY + c]            + at[(size_t)b * N_AY + c];
        const float scale = Cay[(size_t)row * N_AY + D_MODEL + c]  + at[(size_t)b * N_AY + D_MODEL + c];
        const float nrm = (loc[i] - mu) * rsig;
        mod[(size_t)row * D_MODEL + c] = to_bf16(nrm * (1.0f + scale) + shift);
    }
}

// Depthwise causal conv (k=4) + bias + silu over xBC; scatter to xh/Bm/Cm planes.
__global__ void conv_kernel(const float* __restrict__ Czx,
                            const float* __restrict__ conv_w,
                            const float* __restrict__ conv_b,
                            float* __restrict__ xh, float* __restrict__ Bmp,
                            float* __restrict__ Cmp, int total)
{
    int i = blockIdx.x * blockDim.x + threadIdx.x;
    if (i >= total) return;
    const int c = i % CONV_DIM;
    const int l = (i / CONV_DIM) & (SEQ - 1);
    const int b = i / (CONV_DIM * SEQ);
    const size_t rowbase = (size_t)b * SEQ;

    float acc = conv_b[c];
#pragma unroll
    for (int k = 0; k < 4; ++k) {
        const int ll = l + k - 3;
        if (ll >= 0)
            acc += conv_w[c * 4 + k] * Czx[(rowbase + ll) * N_ZX_PAD + D_INNER + c];
    }
    const float v = silu_f(acc);
    const size_t row = rowbase + l;
    if (c < D_INNER)                xh[row * D_INNER + c] = v;
    else if (c < D_INNER + D_STATE) Bmp[row * D_STATE + (c - D_INNER)] = v;
    else                            Cmp[row * D_STATE + (c - D_INNER - D_STATE)] = v;
}

// dt = softplus(raw + dt_bias); dA = exp(dt * (-exp(A_log)))
__global__ void dtda_kernel(const float* __restrict__ Czx,
                            const float* __restrict__ dt_bias,
                            const float* __restrict__ A_log,
                            float* __restrict__ dtv, float* __restrict__ dAb, int total)
{
    int i = blockIdx.x * blockDim.x + threadIdx.x;
    if (i >= total) return;
    const int h = i & (NHEADS - 1);
    const size_t row = (size_t)(i >> 5);
    const float raw = Czx[row * N_ZX_PAD + (D_INNER + CONV_DIM) + h] + dt_bias[h];
    const float dt = (raw > 20.0f) ? raw : log1pf(__expf(raw));
    const float A = -__expf(A_log[h]);
    dtv[i] = dt;
    dAb[i] = __expf(dt * A);
}

// ---------------------------------------------------------------------------
// SSM scan: one block per (b, head). State h[64][128] fp32 held in registers
// (32 floats/thread). Per step: stage B_t/C_t/x_t via LDS, update state,
// contract against C with a 4-lane shuffle reduction (wave32).
// Fuses yv = scan_y + D[h] * xh.
// ---------------------------------------------------------------------------
__global__ __launch_bounds__(256) void scan_kernel(
    const float* __restrict__ xh,  const float* __restrict__ Bm,
    const float* __restrict__ Cm,  const float* __restrict__ dtv,
    const float* __restrict__ dAb, const float* __restrict__ Dp,
    float* __restrict__ Y)
{
    const int b   = blockIdx.x >> 5;
    const int hd  = blockIdx.x & 31;
    const int tid = threadIdx.x;
    const int p   = tid >> 2;        // 0..63  (HEADDIM)
    const int nq  = tid & 3;         // n-quadrant: 32 states each

    __shared__ __align__(16) float sB[D_STATE];
    __shared__ __align__(16) float sC[D_STATE];
    __shared__ __align__(16) float sX[HEADDIM];

    float h[32];
#pragma unroll
    for (int j = 0; j < 32; ++j) h[j] = 0.0f;
    const float Dh = Dp[hd];

    for (int l = 0; l < SEQ; ++l) {
        const size_t row = (size_t)b * SEQ + l;
        if (tid < 128) sB[tid] = Bm[row * D_STATE + tid];
        else           sC[tid - 128] = Cm[row * D_STATE + (tid - 128)];
        if (tid < 64)  sX[tid] = xh[row * D_INNER + hd * HEADDIM + tid];
        __syncthreads();

        const float dA = dAb[row * NHEADS + hd];
        const float dt = dtv[row * NHEADS + hd];
        const float xp = sX[p];
        const float cf = dt * xp;     // dtx

        float y = 0.0f;
        const float4* B4 = reinterpret_cast<const float4*>(sB) + nq * 8;
        const float4* C4 = reinterpret_cast<const float4*>(sC) + nq * 8;
#pragma unroll
        for (int j4 = 0; j4 < 8; ++j4) {
            const float4 bb = B4[j4];
            const float4 cc = C4[j4];
            const int j = j4 * 4;
            h[j]     = h[j]     * dA + cf * bb.x;  y += h[j]     * cc.x;
            h[j + 1] = h[j + 1] * dA + cf * bb.y;  y += h[j + 1] * cc.y;
            h[j + 2] = h[j + 2] * dA + cf * bb.z;  y += h[j + 2] * cc.z;
            h[j + 3] = h[j + 3] * dA + cf * bb.w;  y += h[j + 3] * cc.w;
        }
        y += __shfl_xor(y, 1, 32);
        y += __shfl_xor(y, 2, 32);
        if (nq == 0) Y[row * D_INNER + hd * HEADDIM + p] = y + Dh * xp;
        __syncthreads();
    }
}

// yv_bf16 = (Y) * silu(z)
__global__ void yz_kernel(const float* __restrict__ Y, const float* __restrict__ Czx,
                          bf16* __restrict__ yvb, int total)
{
    int i = blockIdx.x * blockDim.x + threadIdx.x;
    if (i >= total) return;
    const size_t row = (size_t)(i >> 11);
    const int c = i & (D_INNER - 1);
    const float z = Czx[row * N_ZX_PAD + c];
    yvb[i] = to_bf16(Y[i] * silu_f(z));
}

// out = gate * O + x
__global__ void final_kernel(const float* __restrict__ O, const float* __restrict__ Cay,
                             const float* __restrict__ at, const float* __restrict__ x,
                             float* __restrict__ out, int total)
{
    int i = blockIdx.x * blockDim.x + threadIdx.x;
    if (i >= total) return;
    const size_t row = (size_t)(i >> 10);
    const int c = i & (D_MODEL - 1);
    const int b = (int)(row >> 10);
    const float gate = Cay[row * N_AY + 2 * D_MODEL + c] + at[(size_t)b * N_AY + 2 * D_MODEL + c];
    out[i] = gate * O[i] + x[i];
}

// ---------------------------------------------------------------------------
static inline int cdiv(long long a, int b) { return (int)((a + b - 1) / b); }

extern "C" void kernel_launch(void* const* d_in, const int* in_sizes, int n_in,
                              void* d_out, int out_size, void* d_ws, size_t ws_size,
                              hipStream_t stream)
{
    (void)in_sizes; (void)n_in; (void)out_size; (void)ws_size;

    const float* x       = (const float*)d_in[0];
    const float* t       = (const float*)d_in[1];
    const float* y       = (const float*)d_in[2];
    const float* w_ada   = (const float*)d_in[3];
    const float* b_ada   = (const float*)d_in[4];
    const float* w_in    = (const float*)d_in[5];
    const float* conv_w  = (const float*)d_in[6];
    const float* conv_b  = (const float*)d_in[7];
    const float* dt_bias = (const float*)d_in[8];
    const float* A_log   = (const float*)d_in[9];
    const float* D_param = (const float*)d_in[10];
    const float* w_out   = (const float*)d_in[11];
    float* out = (float*)d_out;

    // ---- workspace carve-up ----
    char* ws = (char*)d_ws;
    size_t off = 0;
    auto take = [&](size_t bytes) -> char* {
        char* p = ws + off; off += (bytes + 255) & ~(size_t)255; return p;
    };
    bf16*  bt_ada = (bf16*) take((size_t)N_AY    * D_MODEL * 2);
    bf16*  bt_in  = (bf16*) take((size_t)N_ZX_PAD* D_MODEL * 2);
    bf16*  bt_out = (bf16*) take((size_t)D_MODEL * D_INNER * 2);
    float* at     = (float*)take((size_t)BATCH   * N_AY    * 4);
    bf16*  a_y    = (bf16*) take((size_t)ROWS    * D_MODEL * 2);  // reused as `mod`
    float* Cay    = (float*)take((size_t)ROWS    * N_AY    * 4);
    float* Czx    = (float*)take((size_t)ROWS    * N_ZX_PAD* 4);
    float* xh     = (float*)take((size_t)ROWS    * D_INNER * 4);
    float* Bmp    = (float*)take((size_t)ROWS    * D_STATE * 4);
    float* Cmp    = (float*)take((size_t)ROWS    * D_STATE * 4);
    float* dtv    = (float*)take((size_t)ROWS    * NHEADS  * 4);
    float* dAb    = (float*)take((size_t)ROWS    * NHEADS  * 4);
    float* Yb     = (float*)take((size_t)ROWS    * D_INNER * 4);
    bf16*  yvb    = (bf16*) take((size_t)ROWS    * D_INNER * 2);
    float* Ob     = (float*)take((size_t)ROWS    * D_MODEL * 4);
    bf16*  mod    = a_y;  // a_y fully consumed by the ay GEMM before mod is written

    // ---- 1. weight transpose + bf16 convert (N-major for B-operand layout) ----
    {
        long long n1 = (long long)N_AY * D_MODEL;
        transpose_convert<<<cdiv(n1,256),256,0,stream>>>(w_ada, bt_ada, D_MODEL, N_AY, N_AY, (int)n1);
        long long n2 = (long long)N_ZX_PAD * D_MODEL;
        transpose_convert<<<cdiv(n2,256),256,0,stream>>>(w_in, bt_in, D_MODEL, N_ZX, N_ZX_PAD, (int)n2);
        long long n3 = (long long)D_MODEL * D_INNER;
        transpose_convert<<<cdiv(n3,256),256,0,stream>>>(w_out, bt_out, D_INNER, D_MODEL, D_MODEL, (int)n3);
    }

    // ---- 2. at = silu(t)@w_ada + b_ada ----
    at_kernel<<<cdiv(N_AY,256),256,0,stream>>>(t, w_ada, b_ada, at);

    // ---- 3. a_y = bf16(silu(y)) ----
    {
        long long n = (long long)ROWS * D_MODEL;
        silu_cvt<<<cdiv(n,256),256,0,stream>>>(y, a_y, (int)n);
    }

    // ---- 4. Cay = a_y @ w_ada^T(bf16) + b_ada  (WMMA) ----
    wmma_gemm_bf16<<<dim3(N_AY/128, ROWS/128), 256, 0, stream>>>(
        a_y, bt_ada, Cay, b_ada, D_MODEL, D_MODEL, D_MODEL, N_AY);

    // ---- 5. mod = LN(x)*(1+scale)+shift  -> bf16 ----
    mod_kernel<<<ROWS, 256, 0, stream>>>(x, Cay, at, mod);

    // ---- 6. Czx = mod @ w_in (WMMA, N padded to 4480) ----
    wmma_gemm_bf16<<<dim3(N_ZX_PAD/128, ROWS/128), 256, 0, stream>>>(
        mod, bt_in, Czx, nullptr, D_MODEL, D_MODEL, D_MODEL, N_ZX_PAD);

    // ---- 7. causal depthwise conv + silu -> xh / Bm / Cm ----
    {
        long long n = (long long)ROWS * CONV_DIM;
        conv_kernel<<<cdiv(n,256),256,0,stream>>>(Czx, conv_w, conv_b, xh, Bmp, Cmp, (int)n);
    }

    // ---- 8. dt / dA ----
    {
        long long n = (long long)ROWS * NHEADS;
        dtda_kernel<<<cdiv(n,256),256,0,stream>>>(Czx, dt_bias, A_log, dtv, dAb, (int)n);
    }

    // ---- 9. SSM scan (serial in L, parallel over B*NHEADS=256 blocks) ----
    scan_kernel<<<BATCH*NHEADS, 256, 0, stream>>>(xh, Bmp, Cmp, dtv, dAb, D_param, Yb);

    // ---- 10. yv = Y * silu(z) -> bf16 ----
    {
        long long n = (long long)ROWS * D_INNER;
        yz_kernel<<<cdiv(n,256),256,0,stream>>>(Yb, Czx, yvb, (int)n);
    }

    // ---- 11. O = yv @ w_out (WMMA) ----
    wmma_gemm_bf16<<<dim3(D_MODEL/128, ROWS/128), 256, 0, stream>>>(
        yvb, bt_out, Ob, nullptr, D_INNER, D_INNER, D_INNER, D_MODEL);

    // ---- 12. out = gate * O + x ----
    {
        long long n = (long long)ROWS * D_MODEL;
        final_kernel<<<cdiv(n,256),256,0,stream>>>(Ob, Cay, at, x, out, (int)n);
    }
}
